// QuantumAttention_15728170238729
// MI455X (gfx1250) — matrix operand order repaired and checked by reading, not compiled, not run on hardware
//
#include <hip/hip_runtime.h>
#include <hip/hip_bf16.h>

// ---------------------------------------------------------------------------
// QuantumAttention for gfx1250 (MI455X, wave32, WMMA)
//   S=2048, B=8, D=64
//   out = [ context (S*B*D f32) ; attn_weights (S*B*S f32) ]
// ---------------------------------------------------------------------------

#define QS 2048
#define QB 8
#define QD 64

typedef __attribute__((ext_vector_type(16))) _Float16 v16h;
typedef __attribute__((ext_vector_type(8)))  float    v8f;

union V16H {
    v16h v;
    uint4 q[2];
};

// ---------------------------------------------------------------------------
// Kernel 1: normalize q,k rows; emit f16 hi/lo pairs, layout [i][b][d] (as input)
// ---------------------------------------------------------------------------
__global__ __launch_bounds__(256) void qa_prep_qk(
    const float* __restrict__ q, const float* __restrict__ k,
    _Float16* __restrict__ qhh, _Float16* __restrict__ qhl,
    _Float16* __restrict__ khh, _Float16* __restrict__ khl)
{
    const int NR = QS * QB;                       // 16384 rows per tensor
    int r = blockIdx.x * blockDim.x + threadIdx.x;
    if (r >= 2 * NR) return;
    const float* src = (r < NR) ? q : k;
    _Float16* dh = (r < NR) ? qhh : khh;
    _Float16* dl = (r < NR) ? qhl : khl;
    int row = (r < NR) ? r : (r - NR);

    const float4* s4 = (const float4*)(src + (size_t)row * QD);
    float vals[QD];
    float ss = 0.f;
#pragma unroll
    for (int i = 0; i < QD / 4; ++i) {
        float4 t = s4[i];
        vals[4*i+0] = t.x; vals[4*i+1] = t.y; vals[4*i+2] = t.z; vals[4*i+3] = t.w;
        ss += t.x*t.x + t.y*t.y + t.z*t.z + t.w*t.w;
    }
    float inv = 1.0f / (sqrtf(ss) + 1e-8f);
    _Float16* oh = dh + (size_t)row * QD;
    _Float16* ol = dl + (size_t)row * QD;
#pragma unroll
    for (int d = 0; d < QD; ++d) {
        float xn = vals[d] * inv;
        _Float16 h = (_Float16)xn;
        oh[d] = h;
        ol[d] = (_Float16)(xn - (float)h);
    }
}

// ---------------------------------------------------------------------------
// Kernel 2: V [j][b][d] -> f16 hi/lo transposed [b][d][j]  (row stride QS)
// ---------------------------------------------------------------------------
__global__ __launch_bounds__(256) void qa_prep_v(
    const float* __restrict__ v,
    _Float16* __restrict__ vth, _Float16* __restrict__ vtl)
{
    int t = blockIdx.x * blockDim.x + threadIdx.x;
    if (t >= QS * QB * QD) return;
    int j = t >> 9;            // 8*64 = 512
    int b = (t >> 6) & 7;
    int d = t & 63;
    float val = v[t];
    _Float16 h = (_Float16)val;
    size_t o = ((size_t)(b * QD + d)) * QS + j;
    vth[o] = h;
    vtl[o] = (_Float16)(val - (float)h);
}

// ---------------------------------------------------------------------------
// Kernel 3: fused scores + softmax + attn-out + P*V
// grid: B * (S/16) blocks, 128 threads (4 waves). Wave w owns j-tiles w,w+4,...
// ---------------------------------------------------------------------------
__global__ __launch_bounds__(128) void qa_main(
    const _Float16* __restrict__ qhh, const _Float16* __restrict__ qhl,
    const _Float16* __restrict__ khh, const _Float16* __restrict__ khl,
    const _Float16* __restrict__ vth, const _Float16* __restrict__ vtl,
    float* __restrict__ ctx, float* attn)
{
    __shared__ float sPartial[4][16];
    __shared__ float sInv[16];
    __shared__ float sC[4][16 * QD];

    const int b  = blockIdx.x & 7;
    const int it = blockIdx.x >> 3;
    const int i0 = it * 16;

    const int tid  = threadIdx.x;
    const int w    = tid >> 5;       // wave 0..3
    const int l    = tid & 31;       // lane
    const int n    = l & 15;         // A-row / B-col / C-col index for this lane
    const int half = l >> 4;
    const int c0   = half * 8;       // A-operand K chunk base (split chunks)
    const int kb   = half * 8;       // B-operand K chunk base (split chunks)

    // ---- load A tiles: q_hat rows i0..i0+15, hi/lo, two K-chunks (D=64) ----
    const _Float16* qrh = qhh + ((size_t)(i0 + n) * QB + b) * QD;
    const _Float16* qrl = qhl + ((size_t)(i0 + n) * QB + b) * QD;
    V16H aHi[2], aLo[2];
#pragma unroll
    for (int c = 0; c < 2; ++c) {
        int d0 = 32 * c + c0;
        aHi[c].q[0] = *(const uint4*)(qrh + d0);
        aHi[c].q[1] = *(const uint4*)(qrh + d0 + 16);
        aLo[c].q[0] = *(const uint4*)(qrl + d0);
        aLo[c].q[1] = *(const uint4*)(qrl + d0 + 16);
    }

    // ================= Phase 1: scores, exp, row sums =================
    float rsum[8];
#pragma unroll
    for (int r = 0; r < 8; ++r) rsum[r] = 0.f;

    for (int jt = w; jt < QS / 16; jt += 4) {
        const int j0 = jt * 16;
        // B tile: k_hat rows j0..j0+15 (two split 8-half chunks per lane, d0 and d0+16)
        const _Float16* krh = khh + ((size_t)(j0 + n) * QB + b) * QD;
        const _Float16* krl = khl + ((size_t)(j0 + n) * QB + b) * QD;
        V16H bHi[2], bLo[2];
#pragma unroll
        for (int c = 0; c < 2; ++c) {
            int d0 = 32 * c + kb;
            bHi[c].q[0] = *(const uint4*)(krh + d0);
            bHi[c].q[1] = *(const uint4*)(krh + d0 + 16);
            bLo[c].q[0] = *(const uint4*)(krl + d0);
            bLo[c].q[1] = *(const uint4*)(krl + d0 + 16);
        }

        v8f acc = {};
#pragma unroll
        for (int c = 0; c < 2; ++c) {
            acc = __builtin_amdgcn_wmma_f32_16x16x32_f16(false, aHi[c].v, false, bHi[c].v,
                                                         (short)0, acc, false, false);
            acc = __builtin_amdgcn_wmma_f32_16x16x32_f16(false, aHi[c].v, false, bLo[c].v,
                                                         (short)0, acc, false, false);
            acc = __builtin_amdgcn_wmma_f32_16x16x32_f16(false, aLo[c].v, false, bHi[c].v,
                                                         (short)0, acc, false, false);
        }

#pragma unroll
        for (int r = 0; r < 8; ++r) {
            float cosv = acc[r];
            float ov   = 0.5f + 0.5f * cosv * cosv;
            float e    = __builtin_amdgcn_exp2f(ov * 1.4426950408889634f);
            rsum[r] += e;
            int mr = r + 8 * half;
            attn[((size_t)(i0 + mr) * QB + b) * QS + j0 + n] = e;   // unnormalized
        }
    }

    // reduce row sums over the 16 columns in each half-wave
#pragma unroll
    for (int r = 0; r < 8; ++r) {
        float s = rsum[r];
        s += __shfl_xor(s, 1, 32);
        s += __shfl_xor(s, 2, 32);
        s += __shfl_xor(s, 4, 32);
        s += __shfl_xor(s, 8, 32);
        if (n == 0) sPartial[w][r + 8 * half] = s;
    }

    __threadfence_block();      // make phase-1 global e-stores visible in block
    __syncthreads();
    if (tid < 16) {
        float tot = sPartial[0][tid] + sPartial[1][tid] + sPartial[2][tid] + sPartial[3][tid];
        sInv[tid] = 1.0f / tot;
    }
    __syncthreads();

    // ================= Phase 2: normalize attn + P*V =================
    v8f cacc[4];
#pragma unroll
    for (int t = 0; t < 4; ++t) cacc[t] = (v8f){};

    for (int jc = w; jc < QS / 32; jc += 4) {
        const int jb = jc * 32;
        float* erow = attn + ((size_t)(i0 + n) * QB + b) * QS + jb;
        const float inv = sInv[n];

        // A-operand style gather of the p tile (split K chunks)
        float pv[16];
        {
            float4 t0 = *(const float4*)(erow + c0);
            float4 t1 = *(const float4*)(erow + c0 + 4);
            float4 t2 = *(const float4*)(erow + c0 + 16);
            float4 t3 = *(const float4*)(erow + c0 + 20);
            pv[0]=t0.x*inv; pv[1]=t0.y*inv; pv[2]=t0.z*inv; pv[3]=t0.w*inv;
            pv[4]=t1.x*inv; pv[5]=t1.y*inv; pv[6]=t1.z*inv; pv[7]=t1.w*inv;
            pv[8]=t2.x*inv; pv[9]=t2.y*inv; pv[10]=t2.z*inv; pv[11]=t2.w*inv;
            pv[12]=t3.x*inv; pv[13]=t3.y*inv; pv[14]=t3.z*inv; pv[15]=t3.w*inv;
        }
        V16H pHi, pLo;
#pragma unroll
        for (int u = 0; u < 16; ++u) {
            _Float16 h = (_Float16)pv[u];
            pHi.v[u] = h;
            pLo.v[u] = (_Float16)(pv[u] - (float)h);
        }
        // write the final (normalized) attention weights
        *(float4*)(erow + c0)      = (float4){pv[0], pv[1], pv[2], pv[3]};
        *(float4*)(erow + c0 + 4)  = (float4){pv[4], pv[5], pv[6], pv[7]};
        *(float4*)(erow + c0 + 16) = (float4){pv[8], pv[9], pv[10], pv[11]};
        *(float4*)(erow + c0 + 20) = (float4){pv[12], pv[13], pv[14], pv[15]};

        // P (16x32) * V (32x64) via 4 n-tiles of 16
#pragma unroll
        for (int t = 0; t < 4; ++t) {
            const int d0 = t * 16;
            const _Float16* vrh = vth + ((size_t)(b * QD + d0 + n)) * QS + jb + kb;
            const _Float16* vrl = vtl + ((size_t)(b * QD + d0 + n)) * QS + jb + kb;
            V16H vHi, vLo;
            vHi.q[0] = *(const uint4*)(vrh);
            vHi.q[1] = *(const uint4*)(vrh + 16);
            vLo.q[0] = *(const uint4*)(vrl);
            vLo.q[1] = *(const uint4*)(vrl + 16);
            cacc[t] = __builtin_amdgcn_wmma_f32_16x16x32_f16(false, pHi.v, false, vHi.v,
                                                             (short)0, cacc[t], false, false);
            cacc[t] = __builtin_amdgcn_wmma_f32_16x16x32_f16(false, pHi.v, false, vLo.v,
                                                             (short)0, cacc[t], false, false);
            cacc[t] = __builtin_amdgcn_wmma_f32_16x16x32_f16(false, pLo.v, false, vHi.v,
                                                             (short)0, cacc[t], false, false);
        }
    }

    // stash per-wave context partials, reduce across waves, write context
#pragma unroll
    for (int t = 0; t < 4; ++t)
#pragma unroll
        for (int r = 0; r < 8; ++r) {
            int mr = r + 8 * half;
            sC[w][mr * QD + t * 16 + n] = cacc[t][r];
        }
    __syncthreads();
    for (int idx = tid; idx < 16 * QD; idx += 128) {
        int mr = idx >> 6;
        int d  = idx & 63;
        float s = sC[0][idx] + sC[1][idx] + sC[2][idx] + sC[3][idx];
        ctx[((size_t)(i0 + mr) * QB + b) * QD + d] = s;
    }
}

// ---------------------------------------------------------------------------
// Host launcher
// ---------------------------------------------------------------------------
extern "C" void kernel_launch(void* const* d_in, const int* in_sizes, int n_in,
                              void* d_out, int out_size, void* d_ws, size_t ws_size,
                              hipStream_t stream) {
    const float* q = (const float*)d_in[0];
    const float* k = (const float*)d_in[1];
    const float* v = (const float*)d_in[2];

    float* out  = (float*)d_out;
    float* ctx  = out;                                   // [S,B,D]
    float* attn = out + (size_t)QS * QB * QD;            // [S,B,S]

    const size_t N = (size_t)QS * QB * QD;               // 1,048,576 elements
    _Float16* ws  = (_Float16*)d_ws;                     // 6*N halves = 12 MB
    _Float16* qhh = ws + 0 * N;
    _Float16* qhl = ws + 1 * N;
    _Float16* khh = ws + 2 * N;
    _Float16* khl = ws + 3 * N;
    _Float16* vth = ws + 4 * N;
    _Float16* vtl = ws + 5 * N;

    {
        int rows = 2 * QS * QB;                          // 32768
        qa_prep_qk<<<(rows + 255) / 256, 256, 0, stream>>>(q, k, qhh, qhl, khh, khl);
    }
    {
        int elems = QS * QB * QD;                        // 1,048,576
        qa_prep_v<<<(elems + 255) / 256, 256, 0, stream>>>(v, vth, vtl);
    }
    {
        dim3 grid(QB * (QS / 16));                       // 1024 blocks
        dim3 block(128);                                 // 4 waves (wave32)
        qa_main<<<grid, block, 0, stream>>>(qhh, qhl, khh, khl, vth, vtl, ctx, attn);
    }
}
